// Net_43052752175597
// MI455X (gfx1250) — matrix operand order repaired and checked by reading, not compiled, not run on hardware
//
#include <hip/hip_runtime.h>

#define Nn 10000   // nodes
#define NP 10016   // nodes padded to 32
#define Hh 128     // hidden dim
#define Fk 256     // input feature dim
#define W32 313    // mask words per row (NP/32)
#define LROW 40    // LDS tile row stride in ushorts (64B data + 16B pad)

typedef __attribute__((ext_vector_type(16))) __bf16        v16bf;
typedef __attribute__((ext_vector_type(8)))  float         v8f;
typedef __attribute__((ext_vector_type(8)))  unsigned int  v8u;
typedef __attribute__((ext_vector_type(4)))  unsigned int  u32x4;
typedef __attribute__((ext_vector_type(8)))  int           i32x8;
typedef __attribute__((ext_vector_type(4)))  int           i32x4;

static constexpr size_t PLANE = (size_t)Hh * NP;   // elements per [H][NP] plane

static __device__ __forceinline__ unsigned short bf_rne(float f) {
    unsigned int u = __float_as_uint(f);
    u += 0x7FFFu + ((u >> 16) & 1u);
    return (unsigned short)(u >> 16);
}
static __device__ __forceinline__ float bf_f32(unsigned short h) {
    return __uint_as_float(((unsigned int)h) << 16);
}
static __device__ __forceinline__ unsigned int pack2(unsigned short lo, unsigned short hi) {
    return (unsigned int)lo | ((unsigned int)hi << 16);
}
static __device__ __forceinline__ v16bf cast16(v8u u) {
    return __builtin_bit_cast(v16bf, u);
}
static __device__ __forceinline__ v8f wmma_bf(v16bf a, v16bf b, v8f c) {
    return __builtin_amdgcn_wmma_f32_16x16x32_bf16(false, a, false, b, (short)0, c, false, false);
}
static __device__ __forceinline__ void split_bf(float x, unsigned short& hi, unsigned short& lo) {
    hi = bf_rne(x);
    lo = bf_rne(x - bf_f32(hi));
}
// A-operand K index for VGPR v, lane-half h (16-bit A 16x32 layout)
static __device__ __forceinline__ int kA(int v, int half) {
    return ((v >> 2) << 4) + (half << 3) + ((v & 3) << 1);
}
// B-operand K index for VGPR v, lane-half h (contiguous k = 16h + 2v, used for A and B alike)
static __device__ __forceinline__ int kB(int v, int half) {
    return (half << 4) + (v << 1);
}

// ---------------------------------------------------------------------------
// One-shot prep: W (KxH, fp32) -> W^T hi/lo bf16, f-major [H][K] (paired u32s).
// ---------------------------------------------------------------------------
__global__ __launch_bounds__(256) void prep_w(const float* __restrict__ W, int K,
                                              unsigned short* __restrict__ Thi,
                                              unsigned short* __restrict__ Tlo) {
    const int pairsPerF = K >> 1;
    const int p = blockIdx.x * blockDim.x + threadIdx.x;
    if (p >= Hh * pairsPerF) return;
    const int f = p / pairsPerF;
    const int k = (p - f * pairsPerF) << 1;
    float w0 = W[(size_t)k * Hh + f];
    float w1 = W[(size_t)(k + 1) * Hh + f];
    unsigned short h0, l0, h1, l1;
    split_bf(w0, h0, l0);
    split_bf(w1, h1, l1);
    *(unsigned int*)(Thi + (size_t)f * K + k) = pack2(h0, h1);
    *(unsigned int*)(Tlo + (size_t)f * K + k) = pack2(l0, l1);
}

// ---------------------------------------------------------------------------
// GEMM: hT = W^T (HhxK) @ B^T (Kx10016), split bf16 output [H][NP].
// ---------------------------------------------------------------------------
template <int K, bool B_IS_BF16>
__global__ __launch_bounds__(256) void gemm_fused(const float* __restrict__ Bf32,
                                                  const unsigned short* __restrict__ Bhi16,
                                                  const unsigned short* __restrict__ Blo16,
                                                  const unsigned short* __restrict__ WThi,
                                                  const unsigned short* __restrict__ WTlo,
                                                  unsigned short* __restrict__ hT_hi,
                                                  unsigned short* __restrict__ hT_lo) {
    const int lane = threadIdx.x & 31;
    const int wid  = blockIdx.x * 8 + (threadIdx.x >> 5);
    const int n0   = wid * 32;
    if (n0 >= NP) return;
    const int half = lane >> 4;
    const int l16  = lane & 15;

    const v8f vzero = {0.f, 0.f, 0.f, 0.f, 0.f, 0.f, 0.f, 0.f};
    v8f acc[2][8];
#pragma unroll
    for (int nt = 0; nt < 2; ++nt)
#pragma unroll
        for (int ft = 0; ft < 8; ++ft) acc[nt][ft] = vzero;

    for (int k0 = 0; k0 < K; k0 += 32) {
        v8u bhiU[2], bloU[2];
#pragma unroll
        for (int nt = 0; nt < 2; ++nt) {
            const int n = n0 + nt * 16 + l16;
            if (B_IS_BF16) {
                const size_t base = (size_t)n * K + k0 + (half << 4);
#pragma unroll
                for (int v = 0; v < 8; ++v) {
                    bhiU[nt][v] = *(const unsigned int*)(Bhi16 + base + (v << 1));
                    bloU[nt][v] = *(const unsigned int*)(Blo16 + base + (v << 1));
                }
            } else {
                const int  nc = n < Nn ? n : (Nn - 1);
                const bool ok = n < Nn;
                const float* xp = Bf32 + (size_t)nc * K + k0 + (half << 4);
#pragma unroll
                for (int v = 0; v < 8; ++v) {
                    float2 xv = *(const float2*)(xp + (v << 1));
                    float x0 = ok ? xv.x : 0.f;
                    float x1 = ok ? xv.y : 0.f;
                    unsigned short h0, l0, h1, l1;
                    split_bf(x0, h0, l0);
                    split_bf(x1, h1, l1);
                    bhiU[nt][v] = pack2(h0, h1);
                    bloU[nt][v] = pack2(l0, l1);
                }
            }
        }
        const v16bf Bhi0 = cast16(bhiU[0]), Blo0 = cast16(bloU[0]);
        const v16bf Bhi1 = cast16(bhiU[1]), Blo1 = cast16(bloU[1]);

#pragma unroll
        for (int ft = 0; ft < 8; ++ft) {
            const unsigned short* wah = WThi + (size_t)(ft * 16 + l16) * K + k0;
            const unsigned short* wal = WTlo + (size_t)(ft * 16 + l16) * K + k0;
            v8u ahiU, aloU;
#pragma unroll
            for (int v = 0; v < 8; ++v) {
                const int kl = kB(v, half);
                ahiU[v] = *(const unsigned int*)(wah + kl);
                aloU[v] = *(const unsigned int*)(wal + kl);
            }
            const v16bf Ahi = cast16(ahiU), Alo = cast16(aloU);
            acc[0][ft] = wmma_bf(Ahi, Bhi0, acc[0][ft]);
            acc[0][ft] = wmma_bf(Ahi, Blo0, acc[0][ft]);
            acc[0][ft] = wmma_bf(Alo, Bhi0, acc[0][ft]);
            acc[1][ft] = wmma_bf(Ahi, Bhi1, acc[1][ft]);
            acc[1][ft] = wmma_bf(Ahi, Blo1, acc[1][ft]);
            acc[1][ft] = wmma_bf(Alo, Bhi1, acc[1][ft]);
        }
    }

#pragma unroll
    for (int nt = 0; nt < 2; ++nt) {
        const int n = n0 + nt * 16 + l16;
#pragma unroll
        for (int ft = 0; ft < 8; ++ft)
#pragma unroll
            for (int v = 0; v < 8; ++v) {
                const int f = ft * 16 + (half << 3) + v;   // C/D layout: m=(L/16)*8+v
                unsigned short hi, lo;
                split_bf(acc[nt][ft][v], hi, lo);          // pad nodes -> zeros
                hT_hi[(size_t)f * NP + n] = hi;
                hT_lo[(size_t)f * NP + n] = lo;
            }
    }
}

// ---------------------------------------------------------------------------
// Aggregation partials. Workgroup = 8 waves covering 256 columns that SHARE the
// k-step sequence; the 32x128 hi/lo hT tiles are staged in LDS by the Tensor
// Data Mover (double-buffered, s_wait_tensorcnt + workgroup barriers), cutting
// hT L2 traffic 8x and replacing L2-latency operand loads with ds_load_b128.
// LAYER 1: streams fp32 A once and emits the adjacency bitmask.
// LAYER 2: rebuilds B from the 12.5MB bitmask. K split 8 ways via blockIdx.y.
// ---------------------------------------------------------------------------
template <int LAYER>
__global__ __launch_bounds__(256) void agg_partial(const float* __restrict__ A,
                                                   unsigned int* __restrict__ maskp,
                                                   const unsigned short* __restrict__ hT_hi,
                                                   const unsigned short* __restrict__ hT_lo,
                                                   float* __restrict__ stage) {
    __shared__ uint4 ldsraw[2 * 2 * Hh * (LROW / 8)];   // 2 bufs x 2 planes x 128 rows x 80B = 40KB
    unsigned short* ldsb = (unsigned short*)ldsraw;

    const int lane = threadIdx.x & 31;
    const int wv   = threadIdx.x >> 5;         // wave = node sub-block (32 cols)
    const int n0   = blockIdx.x * 256 + wv * 32;
    const int nw   = n0 >> 5;                  // mask word column
    const bool wok = (n0 < NP);
    const int nwc  = (nw < W32) ? nw : (W32 - 1);
    const int ks   = blockIdx.y;               // K split
    const int half = lane >> 4;
    const int l16  = lane & 15;

    const int s0 = (313 * ks) >> 3;
    const int s1 = (313 * (ks + 1)) >> 3;

    const v8f vzero = {0.f, 0.f, 0.f, 0.f, 0.f, 0.f, 0.f, 0.f};
    v8f acc[2][8];
#pragma unroll
    for (int nt = 0; nt < 2; ++nt)
#pragma unroll
        for (int ft = 0; ft < 8; ++ft) acc[nt][ft] = vzero;

    const int ig0 = n0 + l16;
    const int ig1 = n0 + 16 + l16;
    const float* col0 = A + (ig0 < Nn ? ig0 : Nn - 1);
    const float* col1 = A + (ig1 < Nn ? ig1 : Nn - 1);

    long rowOff[8];
#pragma unroll
    for (int v = 0; v < 8; ++v) rowOff[v] = (long)kB(v, half) * Nn;

    // ---- TDM: issue one 32x128 tile (hi+lo planes) into LDS buffer b --------
    auto issue_tile = [&](int s, int b) {
        const int k0 = s << 5;
#pragma unroll
        for (int p = 0; p < 2; ++p) {
            const unsigned short* src = (p ? hT_lo : hT_hi) + k0;  // row stride NP
            unsigned long long ga = (unsigned long long)(size_t)src;
            // single __shared__ object => block LDS base offset 0
            unsigned lds_addr = (unsigned)(((b << 1) + p) * Hh * (LROW * 2));
            u32x4 g0 = {1u,                                   // count=1
                        lds_addr,
                        (unsigned)ga,
                        (unsigned)((ga >> 32) & 0x1FFFFFFu) | (2u << 30)}; // type=2
            i32x8 g1 = {(1 << 16) | (1 << 20) | (3 << 22) | (3 << 25), // 2B, pad 4dw/16dw
                        (int)((NP & 0xFFFF) << 16),           // tensor_dim0 lo16
                        (Hh << 16),                            // dim0 hi16=0 | tensor_dim1=128
                        (32 << 16),                            // tile_dim0=32
                        Hh,                                    // tile_dim1=128
                        NP,                                    // tensor_dim0_stride lo32
                        0, 0};
            i32x4 g2 = {0, 0, 0, 0};
            i32x4 g3 = {0, 0, 0, 0};
#if __clang_major__ >= 23
            i32x8 g4 = {0, 0, 0, 0, 0, 0, 0, 0};
            __builtin_amdgcn_tensor_load_to_lds(g0, g1, g2, g3, g4, 0);
#else
            __builtin_amdgcn_tensor_load_to_lds(g0, g1, g2, g3, 0);
#endif
        }
    };
    // ---- consume one LDS plane: 16 WMMAs fed by ds_load_b128 ----------------
    auto mac_lds = [&](int b, int p, v16bf B0, v16bf B1) {
        const unsigned short* basep = ldsb + (size_t)(((b << 1) + p) * Hh) * LROW;
#pragma unroll
        for (int ft = 0; ft < 8; ++ft) {
            const unsigned short* row = basep + (ft * 16 + l16) * LROW + (half << 4);
            uint4 c0 = *(const uint4*)(row);
            uint4 c1 = *(const uint4*)(row + 8);
            v8u a;
            a[0] = c0.x; a[1] = c0.y; a[2] = c0.z; a[3] = c0.w;
            a[4] = c1.x; a[5] = c1.y; a[6] = c1.z; a[7] = c1.w;
            const v16bf Av = cast16(a);
            acc[0][ft] = wmma_bf(Av, B0, acc[0][ft]);
            acc[1][ft] = wmma_bf(Av, B1, acc[1][ft]);
        }
    };
    // ---- streamed-operand loads (software-pipelined one step ahead) ---------
    auto load_rawA = [&](int s, float (&a)[8][4]) {
        if (s < 312) {
            const long base = (long)(s << 5) * Nn;
            __builtin_prefetch(col0 + base + rowOff[0] + 128L * Nn, 0, 1);
            __builtin_prefetch(col1 + base + rowOff[0] + 128L * Nn, 0, 1);
#pragma unroll
            for (int v = 0; v < 8; ++v) {
                const long r = base + rowOff[v];
                a[v][0] = col0[r];
                a[v][1] = col0[r + Nn];
                a[v][2] = col1[r];
                a[v][3] = col1[r + Nn];
            }
        } else {                                  // k0=9984: clamp rows (pad bits dont-care)
#pragma unroll
            for (int v = 0; v < 8; ++v) {
                const int j0 = 9984 + kB(v, half);
                const long r0 = (long)(j0 < Nn ? j0 : Nn - 1) * Nn;
                const long r1 = (long)(j0 + 1 < Nn ? j0 + 1 : Nn - 1) * Nn;
                a[v][0] = col0[r0];
                a[v][1] = col0[r1];
                a[v][2] = col1[r0];
                a[v][3] = col1[r1];
            }
        }
    };
    auto load_words = [&](int s, unsigned (&w)[8][2]) {
        const int k0 = s << 5;
#pragma unroll
        for (int v = 0; v < 8; ++v) {
            const int j0 = k0 + kB(v, half);
            w[v][0] = maskp[(size_t)j0 * W32 + nwc];
            w[v][1] = maskp[(size_t)(j0 + 1) * W32 + nwc];
        }
    };
    auto build_B1 = [&](int s, const float (&a)[8][4], v8u& b0u, v8u& b1u) {
        const int k0 = s << 5;
#pragma unroll
        for (int v = 0; v < 8; ++v) {
            const int j0 = k0 + kB(v, half);
            bool e0 = (a[v][0] != 0.f) || (j0 == ig0);
            bool o0 = (a[v][1] != 0.f) || (j0 + 1 == ig0);
            bool e1 = (a[v][2] != 0.f) || (j0 == ig1);
            bool o1 = (a[v][3] != 0.f) || (j0 + 1 == ig1);
            b0u[v] = pack2(e0 ? 0x3F80 : 0, o0 ? 0x3F80 : 0);
            b1u[v] = pack2(e1 ? 0x3F80 : 0, o1 ? 0x3F80 : 0);
            unsigned mE0 = __builtin_amdgcn_ballot_w32(e0);
            unsigned mO0 = __builtin_amdgcn_ballot_w32(o0);
            unsigned mE1 = __builtin_amdgcn_ballot_w32(e1);
            unsigned mO1 = __builtin_amdgcn_ballot_w32(o1);
            unsigned wlo_e = (mE0 & 0xFFFFu) | (mE1 << 16);     // row k0+2v
            unsigned whi_e = (mE0 >> 16) | (mE1 & 0xFFFF0000u); // row k0+16+2v
            unsigned wlo_o = (mO0 & 0xFFFFu) | (mO1 << 16);     // row k0+2v+1
            unsigned whi_o = (mO0 >> 16) | (mO1 & 0xFFFF0000u); // row k0+17+2v
            unsigned wsel = (lane & 2) ? ((lane & 1) ? whi_o : whi_e)
                                       : ((lane & 1) ? wlo_o : wlo_e);
            int rsel = k0 + (v << 1) + (lane & 1) + ((lane & 2) << 3);
            if (wok && lane < 4) maskp[(size_t)rsel * W32 + nw] = wsel;
        }
    };
    auto build_B2 = [&](const unsigned (&w)[8][2], v8u& b0u, v8u& b1u) {
#pragma unroll
        for (int v = 0; v < 8; ++v) {
            b0u[v] = pack2(((w[v][0] >> l16) & 1u) ? 0x3F80 : 0,
                           ((w[v][1] >> l16) & 1u) ? 0x3F80 : 0);
            b1u[v] = pack2(((w[v][0] >> (l16 + 16)) & 1u) ? 0x3F80 : 0,
                           ((w[v][1] >> (l16 + 16)) & 1u) ? 0x3F80 : 0);
        }
    };

    // prologue: first two tiles in flight + first streamed operand
    if (wv == 0) {
        issue_tile(s0, 0);
        if (s0 + 1 < s1) issue_tile(s0 + 1, 1);
    }
    float    araw[8][4];
    unsigned wraw[8][2];
    if (LAYER == 1) load_rawA(s0, araw); else load_words(s0, wraw);

    for (int s = s0; s < s1; ++s) {
        const int b = (s - s0) & 1;
        if (wv == 0) {                             // imm-only builtin: branch on constant
            if (s + 1 < s1) __builtin_amdgcn_s_wait_tensorcnt(2);
            else            __builtin_amdgcn_s_wait_tensorcnt(0);
        }
        __syncthreads();                           // LDS buffer b ready for all waves
        v8u b0u, b1u;
        if (LAYER == 1) build_B1(s, araw, b0u, b1u);
        else            build_B2(wraw, b0u, b1u);
        float    anext[8][4];
        unsigned wnext[8][2];
        if (s + 1 < s1) {
            if (LAYER == 1) load_rawA(s + 1, anext);
            else            load_words(s + 1, wnext);
        }
        const v16bf B0 = cast16(b0u), B1 = cast16(b1u);
        mac_lds(b, 0, B0, B1);                     // hi plane
        mac_lds(b, 1, B0, B1);                     // lo plane
        __syncthreads();                           // all waves done with buffer b
        if (wv == 0 && s + 2 < s1) issue_tile(s + 2, b);
        if (LAYER == 1) {
#pragma unroll
            for (int v = 0; v < 8; ++v)
#pragma unroll
                for (int q = 0; q < 4; ++q) araw[v][q] = anext[v][q];
        } else {
#pragma unroll
            for (int v = 0; v < 8; ++v) {
                wraw[v][0] = wnext[v][0];
                wraw[v][1] = wnext[v][1];
            }
        }
    }

    if (wok) {                                     // deterministic slab store
        float* slab = stage + (size_t)ks * PLANE;
#pragma unroll
        for (int nt = 0; nt < 2; ++nt) {
            const int n = n0 + nt * 16 + l16;
#pragma unroll
            for (int ft = 0; ft < 8; ++ft)
#pragma unroll
                for (int v = 0; v < 8; ++v) {
                    const int f = ft * 16 + (half << 3) + v;
                    slab[(size_t)n * Hh + f] = acc[nt][ft][v];
                }
        }
    }
}

// ---------------------------------------------------------------------------
// Finalize, wave-per-node: coalesced slab reads, shfl-xor sum-of-squares,
// single pass, L2 norm (+ReLU layer1 / double-norm layer2).
// ---------------------------------------------------------------------------
template <int LAYER>
__global__ __launch_bounds__(256) void agg_finalize(const float* __restrict__ stage,
                                                    const float* __restrict__ bias,
                                                    unsigned short* __restrict__ xo_hi,
                                                    unsigned short* __restrict__ xo_lo,
                                                    float* __restrict__ out) {
    const int lane = threadIdx.x & 31;
    const int node = blockIdx.x * 8 + (threadIdx.x >> 5);   // one wave per node
    if (node >= NP) return;
    const size_t roff = (size_t)node * Hh;
    if (node >= Nn) {                 // zero-fill padding for next GEMM's B operand
        if (LAYER == 1) {
#pragma unroll
            for (int j = 0; j < 4; ++j) {
                xo_hi[roff + lane + 32 * j] = 0;
                xo_lo[roff + lane + 32 * j] = 0;
            }
        }
        return;
    }
    float t[4];
#pragma unroll
    for (int j = 0; j < 4; ++j) {
        const int f = lane + 32 * j;
        float s = bias[f];
#pragma unroll
        for (int ks = 0; ks < 8; ++ks) s += stage[(size_t)ks * PLANE + roff + f];
        t[j] = s;
    }
    float ss = t[0] * t[0] + t[1] * t[1] + t[2] * t[2] + t[3] * t[3];
#pragma unroll
    for (int d = 16; d >= 1; d >>= 1) ss += __shfl_xor(ss, d, 32);
    float inv = 1.f / fmaxf(sqrtf(ss), 1e-12f);
    if (LAYER == 2) {
        float n2 = sqrtf(ss) * inv;               // norm after first normalize (~1)
        inv = inv / fmaxf(n2, 1e-12f);            // second F.normalize
    }
#pragma unroll
    for (int j = 0; j < 4; ++j) {
        const int f = lane + 32 * j;
        float val = t[j] * inv;
        if (LAYER == 1) {
            val = fmaxf(val, 0.f);                // ReLU after normalize
            unsigned short hi, lo;
            split_bf(val, hi, lo);
            xo_hi[roff + f] = hi;
            xo_lo[roff + f] = lo;
        } else {
            out[roff + f] = val;
        }
    }
}

// ---------------------------------------------------------------------------
extern "C" void kernel_launch(void* const* d_in, const int* in_sizes, int n_in,
                              void* d_out, int out_size, void* d_ws, size_t ws_size,
                              hipStream_t stream) {
    (void)in_sizes; (void)n_in; (void)out_size; (void)ws_size;
    const float* X  = (const float*)d_in[0];
    const float* A  = (const float*)d_in[1];
    const float* W1 = (const float*)d_in[2];
    const float* b1 = (const float*)d_in[3];
    const float* W2 = (const float*)d_in[4];
    const float* b2 = (const float*)d_in[5];
    float* outp = (float*)d_out;

    unsigned short* ws = (unsigned short*)d_ws;
    unsigned short* hThi  = ws;                         // [H][NP] bf16 (both layers)
    unsigned short* hTlo  = ws + PLANE;
    unsigned short* xhi   = ws + 2 * PLANE;             // [NP][H] bf16 activations
    unsigned short* xlo   = ws + 3 * PLANE;
    unsigned short* wt1hi = ws + 4 * PLANE;             // W1^T split, [H][256]
    unsigned short* wt1lo = wt1hi + (size_t)Hh * Fk;
    unsigned short* wt2hi = wt1lo + (size_t)Hh * Fk;    // W2^T split, [H][128]
    unsigned short* wt2lo = wt2hi + (size_t)Hh * Hh;
    float*        stage = (float*)(wt2lo + (size_t)Hh * Hh);       // 8 slabs [NP][H]
    unsigned int* maskp = (unsigned int*)(stage + 8 * PLANE);      // [NP][W32] bits

    dim3 blk(256);
    prep_w<<<dim3(64), blk, 0, stream>>>(W1, Fk, wt1hi, wt1lo);
    prep_w<<<dim3(32), blk, 0, stream>>>(W2, Hh, wt2hi, wt2lo);
    // layer 1 (streams fp32 A once, emits bitmask)
    gemm_fused<Fk, false><<<dim3(40), blk, 0, stream>>>(X, nullptr, nullptr,
                                                        wt1hi, wt1lo, hThi, hTlo);
    agg_partial<1><<<dim3(40, 8), blk, 0, stream>>>(A, maskp, hThi, hTlo, stage);
    agg_finalize<1><<<dim3(1252), blk, 0, stream>>>(stage, b1, xhi, xlo, nullptr);
    // layer 2 (bitmask only: 12.5MB instead of 400MB)
    gemm_fused<Hh, true><<<dim3(40), blk, 0, stream>>>(nullptr, xhi, xlo,
                                                       wt2hi, wt2lo, hThi, hTlo);
    agg_partial<2><<<dim3(40, 8), blk, 0, stream>>>(A, maskp, hThi, hTlo, stage);
    agg_finalize<2><<<dim3(1252), blk, 0, stream>>>(stage, b2, nullptr, nullptr, outp);
}
